// GraphTransformer_50878182588493
// MI455X (gfx1250) — matrix operand order repaired and checked by reading, not compiled, not run on hardware
//
#include <hip/hip_runtime.h>
#include <hip/hip_bf16.h>

#define NNODES  10000
#define NEDGES  320000
#define BATCH   8
#define SEQ     12
#define HID     64
#define NHEADS  4
#define CDIM    16
#define NLAYERS 3
#define PRED    3

typedef __attribute__((ext_vector_type(16))) _Float16 v16h;
typedef __attribute__((ext_vector_type(8)))  float    v8f;

// ---------------- helpers ----------------
__device__ __forceinline__ unsigned enc_ord(float f) {
    unsigned u = __float_as_uint(f);
    return (u & 0x80000000u) ? ~u : (u | 0x80000000u);
}
__device__ __forceinline__ float dec_ord(unsigned e) {
    return (e & 0x80000000u) ? __uint_as_float(e ^ 0x80000000u)
                             : __uint_as_float(~e);
}

// ---------------- pack weights into WMMA B-fragment layout ----------------
// wpk layout: [l][mat][kk(2)][nt(4)][lane(32)][j(16)]  (f16)
// fragment element j of lane: K = kk*32 + (lane>>4)*8 + (j&7) + (j>=8?16:0) ; N = nt*16 + (lane&15)
__global__ void k_pack_w(const float* __restrict__ Wq, const float* __restrict__ Wk,
                         const float* __restrict__ Wv, const float* __restrict__ Wsk,
                         _Float16* __restrict__ wpk) {
    int t = blockIdx.x * blockDim.x + threadIdx.x;
    const int per = HID * HID;                 // 4096
    if (t >= NLAYERS * 4 * per) return;
    int l  = t / (4 * per);
    int r  = t % (4 * per);
    int mat = r / per;
    int r2  = r % per;
    int kkidx = r2 / 2048;
    int r3    = r2 % 2048;
    int nt    = r3 / 512;
    int r4    = r3 % 512;
    int lane  = r4 / 16;
    int j     = r4 % 16;
    int k = kkidx * 32 + (lane >> 4) * 8 + (j & 7) + ((j >= 8) ? 16 : 0);
    int n = nt * 16 + (lane & 15);
    const float* W = (mat == 0) ? Wq : (mat == 1) ? Wk : (mat == 2) ? Wv : Wsk;
    wpk[t] = (_Float16)W[l * per + k * HID + n];
}

// ---------------- pack h into WMMA A-fragment layout (f32 -> f16) ----------------
// hpk layout per batch: [tile(625)][kk(2)][lane(32)][j(16)]  (f16)
// fragment element j of lane: K = kk*32 + (lane>>4)*8 + (j&7) + (j>=8?16:0)
//                             M = tile*16 + (lane&15)
__global__ void k_pack_a(const float* __restrict__ h, _Float16* __restrict__ hpk) {
    int t = blockIdx.x * blockDim.x + threadIdx.x;
    const int perb = NNODES * HID;             // 640000
    if (t >= BATCH * perb) return;
    int b  = t / perb;
    int r  = t % perb;
    int tile  = r / 1024;
    int r2    = r % 1024;
    int kkidx = r2 / 512;
    int r3    = r2 % 512;
    int lane  = r3 / 16;
    int j     = r3 % 16;
    int k    = kkidx * 32 + (lane >> 4) * 8 + (j & 7) + ((j >= 8) ? 16 : 0);
    int node = tile * 16 + (lane & 15);
    hpk[t] = (_Float16)h[(size_t)b * perb + node * HID + k];
}

// ---------------- input projection: h[b,n,:] = sum_s x[b,s,n]*inW[s,:] + inb ----------------
__global__ void k_input_proj(const float* __restrict__ x, const float* __restrict__ inW,
                             const float* __restrict__ inb, float* __restrict__ h) {
    int t = blockIdx.x * blockDim.x + threadIdx.x;
    if (t >= BATCH * NNODES) return;
    int b = t / NNODES, n = t % NNODES;
    float xs[SEQ];
#pragma unroll
    for (int s = 0; s < SEQ; ++s)
        xs[s] = x[((size_t)b * SEQ + s) * NNODES + n];
    float* hp = h + (size_t)t * HID;
    for (int hh = 0; hh < HID; ++hh) {
        float acc = inb[hh];
#pragma unroll
        for (int s = 0; s < SEQ; ++s) acc = fmaf(xs[s], inW[s * HID + hh], acc);
        hp[hh] = acc;
    }
}

// ---------------- zero fill ----------------
__global__ void k_fill_zero(float* __restrict__ p, int n) {
    int t = blockIdx.x * blockDim.x + threadIdx.x;
    if (t < n) p[t] = 0.0f;
}

// ---------------- WMMA GEMM: O[mat] = A(f16) @ W[mat](f16) + bias[mat] ----------------
// grid = (625 mtiles, 4 matrices), block = 32 (one wave). Each wave: 16 rows x 64 cols.
// Both operands are pre-swizzled: each fragment is one contiguous 32B v16h load.
__global__ void k_gemm_qkvs(const _Float16* __restrict__ Apk, const _Float16* __restrict__ Wpk4,
                            const float* __restrict__ bq, const float* __restrict__ bk,
                            const float* __restrict__ bv, const float* __restrict__ bsk,
                            float* __restrict__ out) {
    const int lane  = threadIdx.x;        // 0..31
    const int mtile = blockIdx.x;         // 0..624
    const int mat   = blockIdx.y;         // 0..3
    const float* bias = (mat == 0) ? bq : (mat == 1) ? bk : (mat == 2) ? bv : bsk;
    const _Float16* Wp = Wpk4 + mat * (HID * HID);      // [kk][nt][lane][16]
    const _Float16* Ab = Apk + (size_t)mtile * 1024;    // [kk][lane][16]
    float* O = out + (size_t)mat * NNODES * HID;

    v8f acc[4];
#pragma unroll
    for (int nt = 0; nt < 4; ++nt) acc[nt] = (v8f){0.f,0.f,0.f,0.f,0.f,0.f,0.f,0.f};

#pragma unroll
    for (int kkidx = 0; kkidx < 2; ++kkidx) {
        v16h afrag = *(const v16h*)(Ab + kkidx * 512 + lane * 16);
#pragma unroll
        for (int nt = 0; nt < 4; ++nt) {
            v16h bfrag = *(const v16h*)(Wp + ((kkidx * 4 + nt) * 32 + lane) * 16);
            acc[nt] = __builtin_amdgcn_wmma_f32_16x16x32_f16(
                false, afrag, false, bfrag, (short)0, acc[nt], false, false);
        }
    }

    const int mrow = lane & 15;
    const int mo   = (lane >> 4) * 8;     // C/D layout: lanes 16-31 hold M = i+8
    const int row0 = mtile * 16;
#pragma unroll
    for (int nt = 0; nt < 4; ++nt) {
        const float bval = bias[nt * 16 + mrow];
#pragma unroll
        for (int i = 0; i < 8; ++i) {
            O[(size_t)(row0 + i + mo) * HID + nt * 16 + mrow] = acc[nt][i] + bval;
        }
    }
}

// ---------------- edge pass 1: logits + segment max (ordered-uint atomicMax) ----------------
__global__ void k_edge_logits_max(const float* __restrict__ q, const float* __restrict__ k,
                                  const int* __restrict__ src, const int* __restrict__ dst,
                                  float* __restrict__ logits, unsigned* __restrict__ m) {
    int t = blockIdx.x * blockDim.x + threadIdx.x;
    if (t >= NEDGES * NHEADS) return;
    int e  = t >> 2;
    int hd = t & 3;
    int s = src[e], d = dst[e];
    const float* qp = q + (size_t)d * HID + hd * CDIM;
    const float* kp = k + (size_t)s * HID + hd * CDIM;
    float dot = 0.f;
#pragma unroll
    for (int c = 0; c < CDIM; ++c) dot = fmaf(qp[c], kp[c], dot);
    float lg = dot * 0.25f;               // 1/sqrt(16)
    logits[t] = lg;
    atomicMax(&m[d * NHEADS + hd], enc_ord(lg));
}

// ---------------- edge pass 2: e = exp(l - m[dst]); den += e; agg += e * v[src] ----------------
__global__ void k_edge_exp_agg(const float* __restrict__ logits, const unsigned* __restrict__ m,
                               const float* __restrict__ v,
                               const int* __restrict__ src, const int* __restrict__ dst,
                               float* __restrict__ den, float* __restrict__ agg) {
    int t = blockIdx.x * blockDim.x + threadIdx.x;
    if (t >= NEDGES * NHEADS) return;
    int e  = t >> 2;
    int hd = t & 3;
    int s = src[e], d = dst[e];
    float mx = dec_ord(m[d * NHEADS + hd]);
    float ex = __expf(logits[t] - mx);
    atomicAdd(&den[d * NHEADS + hd], ex);
    const float* vp = v + (size_t)s * HID + hd * CDIM;
    float* ap = agg + (size_t)d * HID + hd * CDIM;
#pragma unroll
    for (int c = 0; c < CDIM; ++c) atomicAdd(&ap[c], ex * vp[c]);
}

// ---------------- node finish: h = LN(h + agg/den + skip) ; one wave per node ----------------
__global__ void k_node_finish(float* __restrict__ h, const float* __restrict__ agg,
                              const float* __restrict__ den, const float* __restrict__ sk,
                              const float* __restrict__ gam, const float* __restrict__ bet) {
    int n = blockIdx.x;
    int lane = threadIdx.x;               // 0..31, handles c=lane and c=lane+32
    int c0 = lane, c1 = lane + 32;
    size_t base = (size_t)n * HID;

    float dn0 = den[n * NHEADS + (c0 >> 4)];
    float dn1 = den[n * NHEADS + (c1 >> 4)];
    float a0 = (dn0 > 0.f) ? agg[base + c0] / dn0 : 0.f;
    float a1 = (dn1 > 0.f) ? agg[base + c1] / dn1 : 0.f;
    float y0 = h[base + c0] + a0 + sk[base + c0];
    float y1 = h[base + c1] + a1 + sk[base + c1];

    float s1 = y0 + y1, s2 = y0 * y0 + y1 * y1;
#pragma unroll
    for (int off = 16; off > 0; off >>= 1) {
        s1 += __shfl_xor(s1, off, 32);
        s2 += __shfl_xor(s2, off, 32);
    }
    float mu  = s1 * (1.0f / HID);
    float var = s2 * (1.0f / HID) - mu * mu;
    float r   = rsqrtf(var + 1e-5f);
    h[base + c0] = (y0 - mu) * r * gam[c0] + bet[c0];
    h[base + c1] = (y1 - mu) * r * gam[c1] + bet[c1];
}

// ---------------- output projection: out[b,p,n] = h[b,n,:] @ outW[:,p] + outb[p] ----------------
__global__ void k_out_proj(const float* __restrict__ h, const float* __restrict__ oW,
                           const float* __restrict__ ob, float* __restrict__ out) {
    int t = blockIdx.x * blockDim.x + threadIdx.x;
    if (t >= BATCH * NNODES) return;
    int b = t / NNODES, n = t % NNODES;
    const float* hp = h + (size_t)t * HID;
    float a0 = ob[0], a1 = ob[1], a2 = ob[2];
    for (int kx = 0; kx < HID; ++kx) {
        float hv = hp[kx];
        a0 = fmaf(hv, oW[kx * PRED + 0], a0);
        a1 = fmaf(hv, oW[kx * PRED + 1], a1);
        a2 = fmaf(hv, oW[kx * PRED + 2], a2);
    }
    size_t base = (size_t)b * PRED * NNODES + n;
    out[base]               = a0;
    out[base + NNODES]      = a1;
    out[base + 2 * NNODES]  = a2;
}

// ---------------- host launcher ----------------
extern "C" void kernel_launch(void* const* d_in, const int* in_sizes, int n_in,
                              void* d_out, int out_size, void* d_ws, size_t ws_size,
                              hipStream_t stream) {
    const float* x    = (const float*)d_in[0];
    const int*   eidx = (const int*)  d_in[1];
    const float* inW  = (const float*)d_in[2];
    const float* inb  = (const float*)d_in[3];
    const float* Wq   = (const float*)d_in[4];
    const float* bq   = (const float*)d_in[5];
    const float* Wk   = (const float*)d_in[6];
    const float* bk   = (const float*)d_in[7];
    const float* Wv   = (const float*)d_in[8];
    const float* bv   = (const float*)d_in[9];
    const float* Wsk  = (const float*)d_in[10];
    const float* bsk  = (const float*)d_in[11];
    const float* lng  = (const float*)d_in[12];
    const float* lnb  = (const float*)d_in[13];
    const float* oW   = (const float*)d_in[14];
    const float* ob   = (const float*)d_in[15];

    const int* src = eidx;
    const int* dst = eidx + NEDGES;

    // workspace layout (bytes, 256-aligned)
    char* ws = (char*)d_ws;
    float*     h      = (float*)    (ws + 0);          // 20,480,000 B
    _Float16*  hpk    = (_Float16*) (ws + 20480000);   // 10,240,000 B (A fragments)
    _Float16*  wpk    = (_Float16*) (ws + 30720000);   //     98,304 B (B fragments)
    float*     qkvs   = (float*)    (ws + 30818304);   // 10,240,000 B  [4][N][64]
    unsigned*  mmax   = (unsigned*) (ws + 41058304);   //    160,000 B
    float*     den    = (float*)    (ws + 41218304);   //    160,000 B
    float*     agg    = (float*)    (ws + 41378304);   //  2,560,000 B
    float*     logits = (float*)    (ws + 43938304);   //  5,120,000 B  -> end 49,058,304

    const size_t NH = (size_t)NNODES * HID;

    k_pack_w<<<(NLAYERS * 4 * HID * HID + 255) / 256, 256, 0, stream>>>(Wq, Wk, Wv, Wsk, wpk);
    k_input_proj<<<(BATCH * NNODES + 127) / 128, 128, 0, stream>>>(x, inW, inb, h);

    for (int l = 0; l < NLAYERS; ++l) {
        k_pack_a<<<(int)((BATCH * NH + 255) / 256), 256, 0, stream>>>(h, hpk);
        for (int b = 0; b < BATCH; ++b) {
            k_gemm_qkvs<<<dim3(NNODES / 16, 4), 32, 0, stream>>>(
                hpk + (size_t)b * NH, wpk + (size_t)l * 4 * HID * HID,
                bq + l * HID, bk + l * HID, bv + l * HID, bsk + l * HID, qkvs);
            // zero mmax+den+agg (contiguous region: 720,000 dwords)
            k_fill_zero<<<(720000 + 255) / 256, 256, 0, stream>>>((float*)mmax, 720000);
            k_edge_logits_max<<<(NEDGES * NHEADS + 255) / 256, 256, 0, stream>>>(
                qkvs /*q*/, qkvs + NH /*k*/, src, dst, logits, mmax);
            k_edge_exp_agg<<<(NEDGES * NHEADS + 255) / 256, 256, 0, stream>>>(
                logits, mmax, qkvs + 2 * NH /*v*/, src, dst, den, agg);
            k_node_finish<<<NNODES, 32, 0, stream>>>(
                h + (size_t)b * NH, agg, den, qkvs + 3 * NH /*skip*/, lng, lnb);
        }
    }

    k_out_proj<<<(BATCH * NNODES + 127) / 128, 128, 0, stream>>>(h, oW, ob, (float*)d_out);
}
